// MoELayer_67903432950547
// MI455X (gfx1250) — matrix operand order repaired and checked by reading, not compiled, not run on hardware
//
#include <hip/hip_runtime.h>
#include <hip/hip_bf16.h>

// MoE: T=8192, H=1024, F=2048, E=8, top_k=2. fp32 in/out, bf16 WMMA compute.
#define T_TOK 8192
#define H_DIM 1024
#define F_DIM 2048
#define E_NUM 8

typedef __attribute__((ext_vector_type(16))) __bf16 v16bf;
typedef __attribute__((ext_vector_type(8)))  float  v8f;

union Frag { uint4 u[2]; v16bf v; };

__device__ __forceinline__ unsigned short f2bf(float f) {
  unsigned u = __float_as_uint(f);
  u += 0x7FFFu + ((u >> 16) & 1u);          // round-to-nearest-even
  return (unsigned short)(u >> 16);
}

// issue one 32B async global->LDS copy pair (b128 x2); IOFFSET adds to both
// the global and the LDS address per the CDNA5 ISA, so chunk 2 is offset:16.
__device__ __forceinline__ void async_copy32(unsigned lds_off, const void* gptr) {
  unsigned long long ga = (unsigned long long)(size_t)gptr;
  asm volatile(
      "global_load_async_to_lds_b128 %0, %1, off\n\t"
      "global_load_async_to_lds_b128 %0, %1, off offset:16"
      :: "v"(lds_off), "v"(ga) : "memory");
}
__device__ __forceinline__ void wait_async0() {
  asm volatile("s_wait_asynccnt 0x0" ::: "memory");
}

// ---------------- init: zero output, zero per-expert counters ----------------
__global__ void zero_kernel(float* __restrict__ out, int n4, int* __restrict__ cnt) {
  int i = blockIdx.x * blockDim.x + threadIdx.x;
  if (i < n4) ((float4*)out)[i] = make_float4(0.f, 0.f, 0.f, 0.f);
  if (blockIdx.x == 0 && threadIdx.x < E_NUM) cnt[threadIdx.x] = 0;
}

// ---------------- fp32 -> bf16 activation convert ----------------
__global__ void cvt_x_kernel(const float* __restrict__ x, unsigned short* __restrict__ xb, int n4) {
  int i = blockIdx.x * blockDim.x + threadIdx.x;
  if (i >= n4) return;
  float4 v = ((const float4*)x)[i];
  ushort4 o;
  o.x = f2bf(v.x); o.y = f2bf(v.y); o.z = f2bf(v.z); o.w = f2bf(v.w);
  ((ushort4*)xb)[i] = o;
}

// ------- weight repack: in[R][C] fp32, C = A*E interleaved (a*E + e) -------
// out[(e*A + a)*R + r] bf16  (per-expert, N-major, K-contiguous for GEMM B tiles)
__global__ void cvt_w_kernel(const float* __restrict__ in, unsigned short* __restrict__ outw,
                             int R, int C) {
  __shared__ float s[32][33];
  const int r0 = blockIdx.x * 32;
  const int q0 = blockIdx.y * 32;
  const int tx = threadIdx.x & 31, ty = threadIdx.x >> 5;   // 256 threads
  const int A = C / E_NUM;
#pragma unroll
  for (int i = 0; i < 4; ++i) {
    int r = r0 + ty + i * 8;
    s[ty + i * 8][tx] = in[(size_t)r * C + q0 + tx];
  }
  __syncthreads();
#pragma unroll
  for (int i = 0; i < 4; ++i) {
    int q = q0 + ty + i * 8;
    int e = q & (E_NUM - 1), a = q >> 3;
    outw[(size_t)(e * A + a) * R + r0 + tx] = f2bf(s[tx][ty + i * 8]);
  }
}

// ---------------- router: logits, softmax, top-2, dispatch lists ----------------
__global__ void router_kernel(const float* __restrict__ x, const float* __restrict__ rw,
                              int* __restrict__ cnt, int* __restrict__ lists,
                              float* __restrict__ gates) {
  const int wid = threadIdx.x >> 5, lane = threadIdx.x & 31;
  const int t = blockIdx.x * 8 + wid;            // one wave32 per token
  const float* xr = x + (size_t)t * H_DIM;
  float acc[E_NUM];
#pragma unroll
  for (int e = 0; e < E_NUM; ++e) acc[e] = 0.f;
  for (int hh = 0; hh < H_DIM / 32; ++hh) {
    int h = hh * 32 + lane;
    float xv = xr[h];
    const float4* w = (const float4*)(rw + (size_t)h * E_NUM);
    float4 w0 = w[0], w1 = w[1];
    acc[0] += xv * w0.x; acc[1] += xv * w0.y; acc[2] += xv * w0.z; acc[3] += xv * w0.w;
    acc[4] += xv * w1.x; acc[5] += xv * w1.y; acc[6] += xv * w1.z; acc[7] += xv * w1.w;
  }
#pragma unroll
  for (int e = 0; e < E_NUM; ++e)
#pragma unroll
    for (int off = 16; off > 0; off >>= 1)
      acc[e] += __shfl_xor(acc[e], off, 32);
  if (lane == 0) {
    float mx = acc[0];
#pragma unroll
    for (int e = 1; e < E_NUM; ++e) mx = fmaxf(mx, acc[e]);
    float p[E_NUM], se = 0.f;
#pragma unroll
    for (int e = 0; e < E_NUM; ++e) { p[e] = expf(acc[e] - mx); se += p[e]; }
    float inv = 1.f / se;
#pragma unroll
    for (int e = 0; e < E_NUM; ++e) p[e] *= inv;
    int e0 = 0; float p0 = p[0];
#pragma unroll
    for (int e = 1; e < E_NUM; ++e) if (p[e] > p0) { p0 = p[e]; e0 = e; }
    int e1 = -1; float p1 = -1.f;
#pragma unroll
    for (int e = 0; e < E_NUM; ++e) if (e != e0 && p[e] > p1) { p1 = p[e]; e1 = e; }
    int pos0 = atomicAdd(&cnt[e0], 1);
    lists[e0 * T_TOK + pos0] = 2 * t;
    gates[2 * t] = p0;
    int pos1 = atomicAdd(&cnt[e1], 1);
    lists[e1 * T_TOK + pos1] = 2 * t + 1;
    gates[2 * t + 1] = p1;
  }
}

// ---------------- expert GEMM1: h = silu(x @ W1[e]) over gathered rows ----------------
// Block tile 128(M gathered rows) x 128(N=f), K=H step 32, 8 waves each 32x64.
// Double-buffered LDS filled by GLOBAL_LOAD_ASYNC_TO_LDS_B128 (ASYNCcnt).
__global__ __launch_bounds__(256) void gemm1_kernel(
    const unsigned short* __restrict__ xb, const unsigned short* __restrict__ w1t,
    unsigned short* __restrict__ hb, const int* __restrict__ lists,
    const int* __restrict__ cnt) {
  const int e = blockIdx.z;
  const int me = cnt[e];
  if ((int)blockIdx.y * 128 >= me) return;

  __shared__ unsigned short As[2][128][40];   // [buf][M][K], 80B rows (16B aligned)
  __shared__ unsigned short Bs[2][128][40];   // [buf][N][K]
  __shared__ int rid_s[128];

  const int tid = threadIdx.x;
  const int lane = tid & 31, wid = tid >> 5;
  const int half = lane >> 4, l16 = lane & 15;
  const int waveM = wid >> 1, waveN = wid & 1;

  if (tid < 128) {
    int r = blockIdx.y * 128 + tid;
    rid_s[tid] = (r < me) ? lists[e * T_TOK + r] : 0;
  }
  __syncthreads();

  const int n0 = blockIdx.x * 128;
  const int row = tid >> 1, part = tid & 1;
  const int tok = rid_s[row] >> 1;
  const unsigned short* aSrc = xb + (size_t)tok * H_DIM + part * 16;
  const unsigned short* bSrc = w1t + (size_t)(e * F_DIM + n0 + row) * H_DIM + part * 16;
  const unsigned ldsA0 = (unsigned)(size_t)&As[0][row][part * 16];
  const unsigned ldsB0 = (unsigned)(size_t)&Bs[0][row][part * 16];
  const unsigned bufStride = 128u * 40u * 2u;   // bytes per buffer

  v8f acc[2][4];
#pragma unroll
  for (int mt = 0; mt < 2; ++mt)
#pragma unroll
    for (int nt = 0; nt < 4; ++nt)
#pragma unroll
      for (int j = 0; j < 8; ++j) acc[mt][nt][j] = 0.f;

  // prefetch buffer 0
  async_copy32(ldsA0, aSrc);
  async_copy32(ldsB0, bSrc);

  for (int k0 = 0; k0 < H_DIM; k0 += 32) {
    const int buf = (k0 >> 5) & 1;
    wait_async0();        // own wave's fills for `buf` complete
    __syncthreads();      // everyone's fills complete; everyone done with buf^1
    if (k0 + 32 < H_DIM) {
      unsigned off = (buf ^ 1) * bufStride;
      async_copy32(ldsA0 + off, aSrc + k0 + 32);
      async_copy32(ldsB0 + off, bSrc + k0 + 32);
    }

    Frag a[2], b[4];
#pragma unroll
    for (int mt = 0; mt < 2; ++mt) {     // A: K swizzle {0-7,16-23}/{8-15,24-31}
      const unsigned short* p = &As[buf][waveM * 32 + mt * 16 + l16][half * 8];
      a[mt].u[0] = *(const uint4*)p;
      a[mt].u[1] = *(const uint4*)(p + 16);
    }
#pragma unroll
    for (int nt = 0; nt < 4; ++nt) {     // B: K swizzle {0-7,16-23}/{8-15,24-31}
      const unsigned short* p = &Bs[buf][waveN * 64 + nt * 16 + l16][half * 8];
      b[nt].u[0] = *(const uint4*)p;
      b[nt].u[1] = *(const uint4*)(p + 16);
    }
#pragma unroll
    for (int mt = 0; mt < 2; ++mt)
#pragma unroll
      for (int nt = 0; nt < 4; ++nt)
        acc[mt][nt] = __builtin_amdgcn_wmma_f32_16x16x32_bf16(
            false, a[mt].v, false, b[nt].v, (short)0, acc[mt][nt], false, false);
  }

#pragma unroll
  for (int mt = 0; mt < 2; ++mt)
#pragma unroll
    for (int nt = 0; nt < 4; ++nt) {
      int col = n0 + waveN * 64 + nt * 16 + l16;
#pragma unroll
      for (int j = 0; j < 8; ++j) {
        int rib = waveM * 32 + mt * 16 + half * 8 + j;
        if (blockIdx.y * 128 + rib < me) {
          float v = acc[mt][nt][j];
          float s = v / (1.f + expf(-v));         // SiLU fused
          hb[(size_t)rid_s[rib] * F_DIM + col] = f2bf(s);
        }
      }
    }
}

// ---------------- expert GEMM2: out += gate * (h @ W2[e]) ----------------
__global__ __launch_bounds__(256) void gemm2_kernel(
    const unsigned short* __restrict__ hb, const unsigned short* __restrict__ w2t,
    float* __restrict__ out, const int* __restrict__ lists,
    const int* __restrict__ cnt, const float* __restrict__ gates) {
  const int e = blockIdx.z;
  const int me = cnt[e];
  if ((int)blockIdx.y * 128 >= me) return;

  __shared__ unsigned short As[2][128][40];
  __shared__ unsigned short Bs[2][128][40];
  __shared__ int rid_s[128];
  __shared__ float gate_s[128];

  const int tid = threadIdx.x;
  const int lane = tid & 31, wid = tid >> 5;
  const int half = lane >> 4, l16 = lane & 15;
  const int waveM = wid >> 1, waveN = wid & 1;

  if (tid < 128) {
    int r = blockIdx.y * 128 + tid;
    int rid = (r < me) ? lists[e * T_TOK + r] : 0;
    rid_s[tid] = rid;
    gate_s[tid] = gates[rid];
  }
  __syncthreads();

  const int n0 = blockIdx.x * 128;
  const int row = tid >> 1, part = tid & 1;
  const unsigned short* aSrc = hb + (size_t)rid_s[row] * F_DIM + part * 16;
  const unsigned short* bSrc = w2t + (size_t)(e * H_DIM + n0 + row) * F_DIM + part * 16;
  const unsigned ldsA0 = (unsigned)(size_t)&As[0][row][part * 16];
  const unsigned ldsB0 = (unsigned)(size_t)&Bs[0][row][part * 16];
  const unsigned bufStride = 128u * 40u * 2u;

  v8f acc[2][4];
#pragma unroll
  for (int mt = 0; mt < 2; ++mt)
#pragma unroll
    for (int nt = 0; nt < 4; ++nt)
#pragma unroll
      for (int j = 0; j < 8; ++j) acc[mt][nt][j] = 0.f;

  async_copy32(ldsA0, aSrc);
  async_copy32(ldsB0, bSrc);

  for (int k0 = 0; k0 < F_DIM; k0 += 32) {
    const int buf = (k0 >> 5) & 1;
    wait_async0();
    __syncthreads();
    if (k0 + 32 < F_DIM) {
      unsigned off = (buf ^ 1) * bufStride;
      async_copy32(ldsA0 + off, aSrc + k0 + 32);
      async_copy32(ldsB0 + off, bSrc + k0 + 32);
    }

    Frag a[2], b[4];
#pragma unroll
    for (int mt = 0; mt < 2; ++mt) {
      const unsigned short* p = &As[buf][waveM * 32 + mt * 16 + l16][half * 8];
      a[mt].u[0] = *(const uint4*)p;
      a[mt].u[1] = *(const uint4*)(p + 16);
    }
#pragma unroll
    for (int nt = 0; nt < 4; ++nt) {
      const unsigned short* p = &Bs[buf][waveN * 64 + nt * 16 + l16][half * 8];
      b[nt].u[0] = *(const uint4*)p;
      b[nt].u[1] = *(const uint4*)(p + 16);
    }
#pragma unroll
    for (int mt = 0; mt < 2; ++mt)
#pragma unroll
      for (int nt = 0; nt < 4; ++nt)
        acc[mt][nt] = __builtin_amdgcn_wmma_f32_16x16x32_bf16(
            false, a[mt].v, false, b[nt].v, (short)0, acc[mt][nt], false, false);
  }

#pragma unroll
  for (int mt = 0; mt < 2; ++mt)
#pragma unroll
    for (int nt = 0; nt < 4; ++nt) {
      int col = n0 + waveN * 64 + nt * 16 + l16;
#pragma unroll
      for (int j = 0; j < 8; ++j) {
        int rib = waveM * 32 + mt * 16 + half * 8 + j;
        if (blockIdx.y * 128 + rib < me) {
          int rid = rid_s[rib];
          // exactly 2 commutative fp32 adds per element onto zeroed out => bit-deterministic
          atomicAdd(&out[(size_t)(rid >> 1) * H_DIM + col], acc[mt][nt][j] * gate_s[rib]);
        }
      }
    }
}

extern "C" void kernel_launch(void* const* d_in, const int* in_sizes, int n_in,
                              void* d_out, int out_size, void* d_ws, size_t ws_size,
                              hipStream_t stream) {
  const float* x  = (const float*)d_in[0];   // [T,H]
  const float* rw = (const float*)d_in[1];   // [H,E]
  const float* w1 = (const float*)d_in[2];   // [H][F*E] interleaved (f*E + e)
  const float* w2 = (const float*)d_in[3];   // [F][H*E] interleaved (h*E + e)
  float* out = (float*)d_out;

  // workspace layout (~151.4 MB total)
  char* ws = (char*)d_ws;
  unsigned short* xb  = (unsigned short*)ws; ws += (size_t)T_TOK * H_DIM * 2;
  unsigned short* w1t = (unsigned short*)ws; ws += (size_t)E_NUM * F_DIM * H_DIM * 2;
  unsigned short* w2t = (unsigned short*)ws; ws += (size_t)E_NUM * H_DIM * F_DIM * 2;
  unsigned short* hb  = (unsigned short*)ws; ws += (size_t)2 * T_TOK * F_DIM * 2;
  float* gates        = (float*)ws;          ws += (size_t)2 * T_TOK * 4;
  int*   lists        = (int*)ws;            ws += (size_t)E_NUM * T_TOK * 4;
  int*   cnt          = (int*)ws;

  const int n4 = T_TOK * H_DIM / 4;
  zero_kernel<<<(n4 + 255) / 256, 256, 0, stream>>>(out, n4, cnt);
  cvt_x_kernel<<<(n4 + 255) / 256, 256, 0, stream>>>(x, xb, n4);
  // w1: in [H][F*E] -> w1t [(e*F+f)][H]
  cvt_w_kernel<<<dim3(H_DIM / 32, (F_DIM * E_NUM) / 32), 256, 0, stream>>>(w1, w1t, H_DIM, F_DIM * E_NUM);
  // w2: in [F][H*E] -> w2t [(e*H+h)][F]
  cvt_w_kernel<<<dim3(F_DIM / 32, (H_DIM * E_NUM) / 32), 256, 0, stream>>>(w2, w2t, F_DIM, H_DIM * E_NUM);
  router_kernel<<<T_TOK / 8, 256, 0, stream>>>(x, rw, cnt, lists, gates);
  gemm1_kernel<<<dim3(F_DIM / 128, T_TOK / 128, E_NUM), 256, 0, stream>>>(xb, w1t, hb, lists, cnt);
  gemm2_kernel<<<dim3(H_DIM / 128, T_TOK / 128, E_NUM), 256, 0, stream>>>(hb, w2t, out, lists, cnt, gates);
}
